// CausalVQAE_5566277616565
// MI455X (gfx1250) — matrix operand order repaired and checked by reading, not compiled, not run on hardware
//
#include <hip/hip_runtime.h>
#include <hip/hip_bf16.h>

// ---------------------------------------------------------------------------
// CDNA5 (gfx1250) causal VQ-VAE forward, bf16 WMMA with fp32 accumulate.
// Convs are implicit GEMMs on v_wmma_f32_16x16x32_bf16 with 2x2 register
// blocking per wave (32 cout x 32 t, 4 accumulators): 4 WMMAs per K-step at
// half the operand traffic per FLOP of a single-tile wave.
// ---------------------------------------------------------------------------

typedef __bf16 bf16_t;
typedef __attribute__((ext_vector_type(8)))  __bf16 v8bf;
typedef __attribute__((ext_vector_type(16))) __bf16 v16bf;
typedef __attribute__((ext_vector_type(8)))  float  v8f;

#define HALO 160         // zero rows each side: covers 31-col t overshoot * stride
#define CF_LEAKY 1

__device__ __forceinline__ v16bf cat8(v8bf lo, v8bf hi) {
  return __builtin_shufflevector(lo, hi, 0, 1, 2, 3, 4, 5, 6, 7,
                                 8, 9, 10, 11, 12, 13, 14, 15);
}
__device__ __forceinline__ v16bf ldA(const bf16_t* p, int half) {
  return cat8(*(const v8bf*)(p + half * 16), *(const v8bf*)(p + half * 16 + 8));
}
__device__ __forceinline__ v16bf ldB(const bf16_t* p) {
  return cat8(*(const v8bf*)p, *(const v8bf*)(p + 8));
}

// ---------------------------------------------------------------------------
// xb: [B][Ltot][Cip] bf16 (halo'ed, upsample materialized)
// wb: [Coup][K][Cip] bf16 (Coup = ceil32(Cout), zero padded)
// One wave -> 32(cout) x 32(t); 4 waves / block (128 t per block).
// ---------------------------------------------------------------------------
__global__ __launch_bounds__(128) void conv_wmma_bf16(
    const bf16_t* __restrict__ xb, const bf16_t* __restrict__ wb,
    const float* __restrict__ bias, const float* __restrict__ res,
    float* __restrict__ y,
    int Cip, int Cout, int Ltot, int Lout,
    int K, int stride, int dil, int padL, int flags)
{
  const int lane = threadIdx.x & 31;
  const int wv   = threadIdx.x >> 5;
  const int half = lane >> 4;
  const int m16  = lane & 15;
  const int t0   = (blockIdx.x * 4 + wv) * 32;
  if (t0 >= Lout) return;                       // whole-wave uniform exit
  const int co0  = blockIdx.y * 32;
  const int b    = blockIdx.z;

  const bf16_t* __restrict__ xbb = xb + (size_t)b * Ltot * Cip;
  __builtin_prefetch(xbb + (size_t)(t0 * stride + HALO - padL) * Cip, 0, 0);

  const size_t wrow0 = (size_t)(co0 + m16) * K;
  const size_t wrow1 = (size_t)(co0 + 16 + m16) * K;
  const size_t tstep = (size_t)16 * stride * Cip;

  v8f acc[2][2] = {};
  for (int ci0 = 0; ci0 < Cip; ci0 += 32) {
    for (int k = 0; k < K; ++k) {
      const bf16_t* wp0 = wb + (wrow0 + k) * Cip + ci0;
      const bf16_t* wp1 = wb + (wrow1 + k) * Cip + ci0;
      v16bf A0 = ldA(wp0, half);
      v16bf A1 = ldA(wp1, half);
      const int p0 = (t0 + m16) * stride + k * dil - padL + HALO;  // in [0,Ltot)
      const bf16_t* xp0 = xbb + (size_t)p0 * Cip + ci0 + half * 16;
      v16bf B0 = ldB(xp0);
      v16bf B1 = ldB(xp0 + tstep);
      acc[0][0] = __builtin_amdgcn_wmma_f32_16x16x32_bf16(false, A0, false, B0,
                                                          (short)0, acc[0][0], false, false);
      acc[0][1] = __builtin_amdgcn_wmma_f32_16x16x32_bf16(false, A0, false, B1,
                                                          (short)0, acc[0][1], false, false);
      acc[1][0] = __builtin_amdgcn_wmma_f32_16x16x32_bf16(false, A1, false, B0,
                                                          (short)0, acc[1][0], false, false);
      acc[1][1] = __builtin_amdgcn_wmma_f32_16x16x32_bf16(false, A1, false, B1,
                                                          (short)0, acc[1][1], false, false);
    }
  }

  // C/D layout: M = r + 8*half, N = lane%16
#pragma unroll
  for (int ct = 0; ct < 2; ++ct) {
    int t = t0 + ct * 16 + m16;
    if (t >= Lout) continue;
#pragma unroll
    for (int cc = 0; cc < 2; ++cc) {
#pragma unroll
      for (int r = 0; r < 8; ++r) {
        int co = co0 + cc * 16 + r + (half ? 8 : 0);
        if (co < Cout) {
          float v = acc[cc][ct][r] + bias[co];
          size_t oidx = ((size_t)b * Cout + co) * Lout + t;
          if (res) v += res[oidx];
          if (flags & CF_LEAKY) v = v > 0.f ? v : 0.1f * v;
          y[oidx] = v;
        }
      }
    }
  }
}

// ---------------------------------------------------------------------------
// Layout/precision prep kernels (elementwise, L2-resident traffic)
// ---------------------------------------------------------------------------
__global__ void cast_weights(const float* __restrict__ w, bf16_t* __restrict__ wb,
                             int Cout, int Cin, int K, int Coup, int Cip, int flip)
{
  int i = blockIdx.x * blockDim.x + threadIdx.x;
  int total = Coup * K * Cip;
  if (i >= total) return;
  int ci = i % Cip, k = (i / Cip) % K, co = i / (Cip * K);
  float v = 0.f;
  if (co < Cout && ci < Cin)
    v = flip ? w[((size_t)ci * Cout + co) * K + (K - 1 - k)]
             : w[((size_t)co * Cin + ci) * K + k];
  wb[i] = (bf16_t)v;
}

// x [B][Cin][LinPhys] fp32 -> xb [B][Ltot][Cip] bf16, channel-innermost,
// zero halos, nearest-neighbor upsample (factor `up`) materialized.
__global__ void cast_input(const float* __restrict__ x, bf16_t* __restrict__ xb,
                           int B, int Cin, int LinPhys, int up, int Cip, int Ltot)
{
  int i = blockIdx.x * blockDim.x + threadIdx.x;
  int total = B * Ltot * Cip;
  if (i >= total) return;
  int ci = i % Cip, row = (i / Cip) % Ltot, b = i / (Cip * Ltot);
  int p = row - HALO;
  float v = 0.f;
  if (ci < Cin && p >= 0 && p < LinPhys * up)
    v = x[((size_t)b * Cin + ci) * LinPhys + p / up];
  xb[i] = (bf16_t)v;
}

__global__ void cast_plain(const float* __restrict__ x, bf16_t* __restrict__ y, int n)
{
  int i = blockIdx.x * blockDim.x + threadIdx.x;
  if (i < n) y[i] = (bf16_t)x[i];
}

// ---------------------------------------------------------------------------
// VQ: scores[n][c] = R[n,:] . cb[c,:]  (M=row, N=code, K=D) on WMMA
// ---------------------------------------------------------------------------
__global__ __launch_bounds__(32) void vq_scores_bf16(
    const bf16_t* __restrict__ Rb, const bf16_t* __restrict__ cbb,
    float* __restrict__ scores, int N, int Kc, int D)
{
  const int lane = threadIdx.x;
  const int half = lane >> 4, m16 = lane & 15;
  const int c0 = blockIdx.x * 16, n0 = blockIdx.y * 16;

  v8f acc = {};
  const bf16_t* rrow = Rb + (size_t)(n0 + m16) * D;
  const bf16_t* crow = cbb + (size_t)(c0 + m16) * D;
  for (int d0 = 0; d0 < D; d0 += 32) {
    v16bf A = ldA(rrow + d0, half);
    v16bf B = ldB(crow + d0 + half * 16);
    acc = __builtin_amdgcn_wmma_f32_16x16x32_bf16(false, A, false, B,
                                                  (short)0, acc, false, false);
  }
#pragma unroll
  for (int r = 0; r < 8; ++r) {
    int n = n0 + r + (half ? 8 : 0);
    if (n < N) scores[(size_t)n * Kc + (c0 + m16)] = acc[r];
  }
}

__global__ void vq_cnorm(const float* __restrict__ cb, float* __restrict__ cnorm,
                         int total, int D)
{
  int i = blockIdx.x * blockDim.x + threadIdx.x;
  if (i >= total) return;
  const float* r = cb + (size_t)i * D;
  float s = 0.f;
  for (int d = 0; d < D; ++d) s += r[d] * r[d];
  cnorm[i] = s;
}

__global__ void vq_argmin(const float* __restrict__ scores,
                          const float* __restrict__ cnorm,
                          int* __restrict__ idx, int Kc)
{
  __shared__ float sv[256];
  __shared__ int   si[256];
  const int n = blockIdx.x, tid = threadIdx.x;
  float best = 3.4e38f; int bi = 0;
  for (int c = tid; c < Kc; c += blockDim.x) {
    float d = cnorm[c] - 2.f * scores[(size_t)n * Kc + c];
    if (d < best || (d == best && c < bi)) { best = d; bi = c; }
  }
  sv[tid] = best; si[tid] = bi;
  __syncthreads();
  for (int s = 128; s > 0; s >>= 1) {
    if (tid < s) {
      if (sv[tid + s] < sv[tid] ||
          (sv[tid + s] == sv[tid] && si[tid + s] < si[tid])) {
        sv[tid] = sv[tid + s]; si[tid] = si[tid + s];
      }
    }
    __syncthreads();
  }
  if (tid == 0) idx[n] = si[0];
}

__global__ void vq_update(float* __restrict__ R, float* __restrict__ qsum,
                          const float* __restrict__ cb, const int* __restrict__ idx,
                          float* __restrict__ commit, float* __restrict__ idx_out,
                          int D, int q, int NQ, float scale)
{
  __shared__ float sv[256];
  const int n = blockIdx.x, tid = threadIdx.x;
  const int code = idx[n];
  float local = 0.f;
  for (int d = tid; d < D; d += blockDim.x) {
    float qv = cb[(size_t)code * D + d];
    float df = R[(size_t)n * D + d] - qv;
    local += df * df;
    R[(size_t)n * D + d] = df;
    qsum[(size_t)n * D + d] += qv;
  }
  sv[tid] = local;
  __syncthreads();
  for (int s = 128; s > 0; s >>= 1) {
    if (tid < s) sv[tid] += sv[tid + s];
    __syncthreads();
  }
  if (tid == 0) {
    atomicAdd(commit, sv[0] * scale);
    idx_out[(size_t)n * NQ + q] = (float)code;
  }
}

__global__ void nch_to_ntc(const float* __restrict__ z, float* __restrict__ R,
                           int B, int C, int T)
{
  int i = blockIdx.x * blockDim.x + threadIdx.x;
  if (i >= B * C * T) return;
  int t = i % T, c = (i / T) % C, b = i / (T * C);
  R[((size_t)b * T + t) * C + c] = z[i];
}
__global__ void ntc_to_nch(const float* __restrict__ R, float* __restrict__ z,
                           int B, int C, int T)
{
  int i = blockIdx.x * blockDim.x + threadIdx.x;
  if (i >= B * C * T) return;
  int t = i % T, c = (i / T) % C, b = i / (T * C);
  z[i] = R[((size_t)b * T + t) * C + c];
}

// ---------------------------------------------------------------------------
// Host orchestration
// ---------------------------------------------------------------------------
extern "C" void kernel_launch(void* const* d_in, const int* in_sizes, int n_in,
                              void* d_out, int out_size, void* d_ws, size_t ws_size,
                              hipStream_t stream)
{
  (void)in_sizes; (void)n_in; (void)out_size; (void)ws_size;
  const int Bsz = 4, L0 = 48000, D = 512, NQ = 8, KC = 1024, T = 100, N = Bsz * T;
  auto F = [&](int i) { return (const float*)d_in[i]; };

  // JAX pytree flatten (sorted dict keys):
  // 0: codebooks
  // dec block j: base = 1 + 14j : res{0..2}.{c1,c2}.{b,w} (12), up.b, up.w
  // 71: dec_in.b  72: dec_in.w  73: dec_out.b  74: dec_out.w
  // enc block j: base = 75 + 14j : down.b, down.w, res{0..2}.{c1,c2}.{b,w}
  // 145: enc_in.b 146: enc_in.w 147: enc_out.b 148: enc_out.w 149: x

  const size_t CAP = 6 * 1024 * 1024;       // fp32 ping-pong buffers
  float* bufA   = (float*)d_ws;
  float* bufB   = bufA + CAP;
  float* bufC   = bufB + CAP;
  float* Rres   = bufC + CAP;               // [N,D]
  float* qsum   = Rres + (size_t)N * D;
  float* scores = qsum + (size_t)N * D;     // [N,KC]
  float* cnorm  = scores + (size_t)N * KC;  // [NQ,KC]
  int*   idxbuf = (int*)(cnorm + (size_t)NQ * KC);
  bf16_t* wbf = (bf16_t*)(idxbuf + 512);    // <= 512*11*1024 elems
  bf16_t* xbf = wbf + 6000000;              // <= ~12.5M elems (decoder up-conv inputs)
  bf16_t* cbb = xbf + 6700000;              // 8*1024*512
  bf16_t* Rb  = cbb + (size_t)NQ * KC * D;  // [N,D]

  float* out    = (float*)d_out;            // [B*L0]
  float* commit = out + (size_t)Bsz * L0;
  float* idxout = commit + 1;               // [N,NQ] as float

  auto conv = [&](const float* x, int wleaf, int bleaf, const float* res, float* y,
                  int Cin, int Cout, int LinPhys, int up, int Lout, int K,
                  int stride, int dil, int padL, int flags, int flip) {
    int Cip  = (Cin + 31) & ~31;
    int Coup = (Cout + 31) & ~31;            // pad to 32 for 2x2 blocking
    int Ltot = LinPhys * up + 2 * HALO;
    int wt = Coup * K * Cip;
    cast_weights<<<(wt + 255) / 256, 256, 0, stream>>>(F(wleaf), wbf, Cout, Cin,
                                                       K, Coup, Cip, flip);
    int xt = Bsz * Ltot * Cip;
    cast_input<<<(xt + 255) / 256, 256, 0, stream>>>(x, xbf, Bsz, Cin, LinPhys,
                                                     up, Cip, Ltot);
    dim3 g((Lout + 127) / 128, Coup / 32, Bsz);
    conv_wmma_bf16<<<g, 128, 0, stream>>>(xbf, wbf, F(bleaf), res, y, Cip, Cout,
                                          Ltot, Lout, K, stride, dil, padL, flags);
  };
  auto swp = [](float*& a, float*& b) { float* t = a; a = b; b = t; };

  const int strides[5] = {2, 3, 4, 4, 5};
  const int dils[3]    = {1, 3, 9};

  // ------------------------- Encoder -------------------------
  int L = L0, ch = 32;
  conv(F(149), 146, 145, nullptr, bufA, 1, 32, L, 1, L, 7, 1, 1, 6, 0, 0);
  for (int blk = 0; blk < 5; ++blk) {
    int base = 75 + 14 * blk;
    for (int rb = 0; rb < 3; ++rb) {
      int d = dils[rb];
      int c1b = base + 2 + 4 * rb, c1w = c1b + 1, c2b = c1b + 2, c2w = c1b + 3;
      conv(bufA, c1w, c1b, nullptr, bufB, ch, ch, L, 1, L, 7, 1, d, 6 * d, CF_LEAKY, 0);
      conv(bufB, c2w, c2b, bufA, bufC, ch, ch, L, 1, L, 1, 1, 1, 0, CF_LEAKY, 0);
      swp(bufA, bufC);
    }
    int s = strides[blk];
    conv(bufA, base + 1, base + 0, nullptr, bufB, ch, ch * 2, L, 1, L / s,
         2 * s + 1, s, 1, s + 1, CF_LEAKY, 0);
    swp(bufA, bufB);
    ch *= 2; L /= s;
  }
  conv(bufA, 148, 147, nullptr, bufB, 1024, D, T, 1, T, 3, 1, 1, 2, 0, 0);

  // ------------------------- Residual VQ -------------------------
  {
    int total = Bsz * D * T;
    nch_to_ntc<<<(total + 255) / 256, 256, 0, stream>>>(bufB, Rres, Bsz, D, T);
    hipMemsetAsync(qsum, 0, (size_t)N * D * sizeof(float), stream);
    hipMemsetAsync(commit, 0, sizeof(float), stream);
    vq_cnorm<<<(NQ * KC + 127) / 128, 128, 0, stream>>>(F(0), cnorm, NQ * KC, D);
    int cbn = NQ * KC * D;
    cast_plain<<<(cbn + 255) / 256, 256, 0, stream>>>(F(0), cbb, cbn);
    for (int q = 0; q < NQ; ++q) {
      cast_plain<<<(N * D + 255) / 256, 256, 0, stream>>>(Rres, Rb, N * D);
      dim3 g(KC / 16, N / 16);
      vq_scores_bf16<<<g, 32, 0, stream>>>(Rb, cbb + (size_t)q * KC * D, scores,
                                           N, KC, D);
      vq_argmin<<<N, 256, 0, stream>>>(scores, cnorm + (size_t)q * KC, idxbuf, KC);
      vq_update<<<N, 256, 0, stream>>>(Rres, qsum, F(0) + (size_t)q * KC * D,
                                       idxbuf, commit, idxout, D, q, NQ,
                                       1.f / (float)(N * D));
    }
    ntc_to_nch<<<(total + 255) / 256, 256, 0, stream>>>(qsum, bufA, Bsz, D, T);
  }

  // ------------------------- Decoder -------------------------
  conv(bufA, 72, 71, nullptr, bufB, D, 1024, T, 1, T, 7, 1, 1, 6, 0, 1);
  swp(bufA, bufB);
  int ch2 = 1024, L2 = T;
  const int dstr[5] = {5, 4, 4, 3, 2};
  for (int j = 0; j < 5; ++j) {
    int base = 1 + 14 * j;
    int s = dstr[j];
    conv(bufA, base + 13, base + 12, nullptr, bufB, ch2, ch2 / 2, L2, s, L2 * s,
         2 * s + 1, 1, 1, s, CF_LEAKY, 0);
    swp(bufA, bufB);
    ch2 /= 2; L2 *= s;
    for (int rb = 0; rb < 3; ++rb) {
      int d = dils[rb];
      int c1b = base + 4 * rb, c1w = c1b + 1, c2b = c1b + 2, c2w = c1b + 3;
      conv(bufA, c1w, c1b, nullptr, bufB, ch2, ch2, L2, 1, L2, 7, 1, d, 6 * d,
           CF_LEAKY, 0);
      conv(bufB, c2w, c2b, bufA, bufC, ch2, ch2, L2, 1, L2, 1, 1, 1, 0,
           CF_LEAKY, 0);
      swp(bufA, bufC);
    }
  }
  conv(bufA, 74, 73, nullptr, out, 32, 1, L0, 1, L0, 7, 1, 1, 6, 0, 0);
}
